// BiDirectionalCrossAttention_163208757336
// MI455X (gfx1250) — matrix operand order repaired and checked by reading, not compiled, not run on hardware
//
#include <hip/hip_runtime.h>
#include <hip/hip_bf16.h>

typedef __attribute__((ext_vector_type(16))) __bf16 v16bf;
typedef __attribute__((ext_vector_type(8)))  __bf16 v8bf;
typedef __attribute__((ext_vector_type(4)))  __bf16 v4bf;
typedef __attribute__((ext_vector_type(8)))  float  v8f;
typedef __attribute__((ext_vector_type(4)))  float  v4f;

#define B_N  32768
#define DIM  512
#define H    8
#define DHD  64
#define LDSP 520   // padded bf16 row stride: 1040 B, 16B-aligned, stride/4 = 260 -> distinct bank groups

__device__ __forceinline__ v16bf cat16(v8bf lo, v8bf hi) {
  v16bf r;
#pragma unroll
  for (int i = 0; i < 8; ++i) { r[i] = lo[i]; r[i + 8] = hi[i]; }
  return r;
}

// One wave computes a 16x64 f32 strip: 4 adjacent 16x16 tiles sharing one A
// fragment per K-step.  C[:, n0+64) = A[r0:r0+16, :] * W[n0:n0+64, :]^T
// A, W bf16 row-major, ld = DIM; K = DIM via v_wmma_f32_16x16x32_bf16.
__device__ __forceinline__ void wave_gemm_strip4(const __bf16* __restrict__ A,
                                                 const __bf16* __restrict__ W,
                                                 int r0, int n0, int lane,
                                                 v8f acc[4]) {
  const int lane16 = lane & 15;
  const int khalf  = lane >> 4;
  const __bf16* arow = A + (size_t)(r0 + lane16) * DIM;
  const __bf16* wrow = W + (size_t)(n0 + lane16) * DIM;
#pragma unroll 2
  for (int k0 = 0; k0 < DIM; k0 += 32) {
    // A fragment: lanes 0-15 -> K[k0+0..7] & K[k0+16..23]; lanes 16-31 -> +8
    const v8bf* ap = (const v8bf*)(arow + k0 + khalf * 8);
    v16bf a = cat16(ap[0], ap[2]);
#pragma unroll
    for (int t = 0; t < 4; ++t) {
      // B fragment tile t: lanes 0-15 -> K[k0+0..7] & K[k0+16..23]; lanes 16-31 -> +8
      const v8bf* bp = (const v8bf*)(wrow + (size_t)t * 16 * DIM + k0 + khalf * 8);
      v16bf b = cat16(bp[0], bp[2]);
      acc[t] = __builtin_amdgcn_wmma_f32_16x16x32_bf16(false, a, false, b,
                                                       (short)0, acc[t], false, false);
    }
  }
}

__global__ __launch_bounds__(256)
void cvt_f32_bf16(const float* __restrict__ src, __bf16* __restrict__ dst, int n4) {
  int i = blockIdx.x * 256 + threadIdx.x;
  if (i < n4) {
    v4f v = ((const v4f*)src)[i];
    v4bf o;
#pragma unroll
    for (int j = 0; j < 4; ++j) o[j] = (__bf16)v[j];
    ((v4bf*)dst)[i] = o;
  }
}

// One block = 16 rows of B. Computes q,k,v for one direction into LDS via WMMA,
// runs the (head-mixing) 8x8 attention per sample, writes out bf16; repeats for
// the second direction reusing the same LDS.
__global__ __launch_bounds__(256)
void fused_qkv_attn(const __bf16* __restrict__ xP, const __bf16* __restrict__ xL,
                    const __bf16* __restrict__ wbf,   // q1,k1,v1,q2,k2,v2 each DIM*DIM
                    const float* __restrict__ bq1, const float* __restrict__ bk1,
                    const float* __restrict__ bv1, const float* __restrict__ bq2,
                    const float* __restrict__ bk2, const float* __restrict__ bv2,
                    __bf16* __restrict__ out1, __bf16* __restrict__ out2) {
  __shared__ __bf16 Qs[16 * LDSP];
  __shared__ __bf16 Ks[16 * LDSP];
  __shared__ __bf16 Vs[16 * LDSP];
  __shared__ float  Sc[16][H][H];

  const int tid    = threadIdx.x;
  const int wave   = tid >> 5;
  const int lane   = tid & 31;
  const int lane16 = lane & 15;
  const int khalf  = lane >> 4;
  const int r0     = blockIdx.x * 16;

  for (int dir = 0; dir < 2; ++dir) {
    const __bf16* Aq  = dir ? xL : xP;
    const __bf16* Akv = dir ? xP : xL;
    const __bf16* Wq  = wbf + (size_t)(dir * 3 + 0) * DIM * DIM;
    const __bf16* Wk  = wbf + (size_t)(dir * 3 + 1) * DIM * DIM;
    const __bf16* Wv  = wbf + (size_t)(dir * 3 + 2) * DIM * DIM;
    const float*  bq  = dir ? bq2 : bq1;
    const float*  bk  = dir ? bk2 : bk1;
    const float*  bv  = dir ? bv2 : bv1;

    // 3 matrices x 8 quad-strips (64 cols each) = 24 jobs over 8 waves
    for (int qj = wave; qj < 24; qj += 8) {
      const int mat = qj >> 3;              // 0=q 1=k 2=v
      const int n0  = (qj & 7) << 6;
      const __bf16* A  = (mat == 0) ? Aq : Akv;
      const __bf16* W  = (mat == 0) ? Wq : (mat == 1 ? Wk : Wv);
      const float*  bb = (mat == 0) ? bq : (mat == 1 ? bk : bv);
      __bf16* dst      = (mat == 0) ? Qs : (mat == 1 ? Ks : Vs);

      v8f acc[4] = {v8f{}, v8f{}, v8f{}, v8f{}};
      wave_gemm_strip4(A, W, r0, n0, lane, acc);

      const int m0 = khalf * 8;             // C layout: vgpr j -> row m0+j
#pragma unroll
      for (int t = 0; t < 4; ++t) {
        const int col  = n0 + t * 16 + lane16;
        const float bias = bb[col];
#pragma unroll
        for (int j = 0; j < 8; ++j)
          dst[(m0 + j) * LDSP + col] = (__bf16)(acc[t][j] + bias);
      }
    }
    __syncthreads();

    // scores[s][h][e] = q[s,h,:] . k[s,e,:] / sqrt(DH)   (16 threads per sample)
    {
      const int s   = tid >> 4;
      const int sub = tid & 15;
#pragma unroll
      for (int c = 0; c < 4; ++c) {
        const int flat = sub * 4 + c;
        const int h = flat >> 3, e = flat & 7;
        const __bf16* qp = Qs + s * LDSP + h * DHD;
        const __bf16* kp = Ks + s * LDSP + e * DHD;
        float a = 0.f;
#pragma unroll 8
        for (int d = 0; d < DHD; ++d) a += (float)qp[d] * (float)kp[d];
        Sc[s][h][e] = a * 0.125f;
      }
    }
    __syncthreads();

    if (tid < 128) {  // softmax over e for each (sample, head)
      const int s = tid >> 3, h = tid & 7;
      float m = Sc[s][h][0];
#pragma unroll
      for (int e = 1; e < 8; ++e) m = fmaxf(m, Sc[s][h][e]);
      float ex[8], sum = 0.f;
#pragma unroll
      for (int e = 0; e < 8; ++e) { ex[e] = __expf(Sc[s][h][e] - m); sum += ex[e]; }
      const float inv = 1.f / sum;
#pragma unroll
      for (int e = 0; e < 8; ++e) Sc[s][h][e] = ex[e] * inv;
    }
    __syncthreads();

    {  // out[s, h*64+d] = sum_e attn[s,h,e] * v[s, e*64+d]  -> global bf16
      __bf16* outg = dir ? out2 : out1;
      const int s  = tid >> 4;
      const int c0 = (tid & 15) * 32;
#pragma unroll
      for (int c = 0; c < 32; ++c) {
        const int col = c0 + c;
        const int h = col >> 6, dd = col & 63;
        float a = 0.f;
#pragma unroll
        for (int e = 0; e < 8; ++e)
          a += Sc[s][h][e] * (float)Vs[s * LDSP + e * DHD + dd];
        outg[(size_t)(r0 + s) * DIM + col] = (__bf16)a;
      }
    }
    __syncthreads();
  }
}

// Output projections: d_out[which][r,n] = out_bf[r,:] . Wo[n,:] + bo[n]
// 8 waves x one 16x64 strip each = full 512 columns.
__global__ __launch_bounds__(256)
void out_proj(const __bf16* __restrict__ o1, const __bf16* __restrict__ o2,
              const __bf16* __restrict__ Wo1, const __bf16* __restrict__ Wo2,
              const float* __restrict__ bo1, const float* __restrict__ bo2,
              float* __restrict__ dst) {
  const int which  = blockIdx.y;
  const __bf16* A  = which ? o2 : o1;
  const __bf16* W  = which ? Wo2 : Wo1;
  const float*  bb = which ? bo2 : bo1;
  float* out = dst + (size_t)which * B_N * DIM;

  const int tid    = threadIdx.x;
  const int wave   = tid >> 5;
  const int lane   = tid & 31;
  const int lane16 = lane & 15;
  const int khalf  = lane >> 4;
  const int r0     = blockIdx.x * 16;
  const int n0     = wave << 6;

  v8f acc[4] = {v8f{}, v8f{}, v8f{}, v8f{}};
  wave_gemm_strip4(A, W, r0, n0, lane, acc);

  const int m0 = khalf * 8;
#pragma unroll
  for (int t = 0; t < 4; ++t) {
    const int col  = n0 + t * 16 + lane16;
    const float bias = bb[col];
#pragma unroll
    for (int j = 0; j < 8; ++j)
      out[(size_t)(r0 + m0 + j) * DIM + col] = acc[t][j] + bias;
  }
}

extern "C" void kernel_launch(void* const* d_in, const int* in_sizes, int n_in,
                              void* d_out, int out_size, void* d_ws, size_t ws_size,
                              hipStream_t stream) {
  const float* xPf = (const float*)d_in[0];
  const float* xLf = (const float*)d_in[1];
  const float* Wf[8];
  const float* bf_[8];
  for (int i = 0; i < 8; ++i) {           // order: q1,k1,v1,q2,k2,v2,o1,o2
    Wf[i]  = (const float*)d_in[2 + 2 * i];
    bf_[i] = (const float*)d_in[3 + 2 * i];
  }

  __bf16* ws = (__bf16*)d_ws;
  const size_t WE = (size_t)DIM * DIM;    // 262144
  const size_t XE = (size_t)B_N * DIM;    // 16777216
  __bf16* wbf = ws;                       // 8 weight matrices (bf16)
  __bf16* xPb = ws + 8 * WE;
  __bf16* xLb = xPb + XE;
  __bf16* o1b = xLb + XE;
  __bf16* o2b = o1b + XE;

  for (int i = 0; i < 8; ++i)
    cvt_f32_bf16<<<(int)(WE / 4 / 256), 256, 0, stream>>>(Wf[i], wbf + i * WE, (int)(WE / 4));
  cvt_f32_bf16<<<(int)(XE / 4 / 256), 256, 0, stream>>>(xPf, xPb, (int)(XE / 4));
  cvt_f32_bf16<<<(int)(XE / 4 / 256), 256, 0, stream>>>(xLf, xLb, (int)(XE / 4));

  fused_qkv_attn<<<B_N / 16, 256, 0, stream>>>(xPb, xLb, wbf,
      bf_[0], bf_[1], bf_[2], bf_[3], bf_[4], bf_[5], o1b, o2b);

  out_proj<<<dim3(B_N / 16, 2), 256, 0, stream>>>(o1b, o2b,
      wbf + 6 * WE, wbf + 7 * WE, bf_[6], bf_[7], (float*)d_out);
}
